// MultiHeadAttention_14989435863278
// MI455X (gfx1250) — matrix operand order repaired and checked by reading, not compiled, not run on hardware
//
#include <hip/hip_runtime.h>

typedef _Float16 v16h __attribute__((ext_vector_type(16)));
typedef _Float16 v8h  __attribute__((ext_vector_type(8)));
typedef float    v8f  __attribute__((ext_vector_type(8)));

constexpr int   kB   = 2;
constexpr int   kS   = 2048;
constexpr int   kD   = 1024;
constexpr int   kH   = 16;
constexpr int   kHD  = 64;
constexpr int   kM   = kB * kS;          // 4096 flattened rows
constexpr float kNegInf = -3.0e38f;

__device__ __forceinline__ v8f wmma_f16(v16h a, v16h b, v8f c) {
  // D = A(16x32 f16) * B(32x16 f16) + C(16x16 f32)
  return __builtin_amdgcn_wmma_f32_16x16x32_f16(
      /*neg_a=*/false, a, /*neg_b=*/false, b,
      /*c_mod=*/(short)0, c, /*reuse_a=*/false, /*reuse_b=*/false);
}

// Max-reduction across each 16-lane half using DPP (VALU-only butterfly).
__device__ __forceinline__ float half16_max(float x) {
#if __has_builtin(__builtin_amdgcn_update_dpp)
  int t;
  t = __builtin_amdgcn_update_dpp(0, __float_as_int(x), 0xB1, 0xF, 0xF, true);
  x = fmaxf(x, __int_as_float(t));   // lane ^ 1
  t = __builtin_amdgcn_update_dpp(0, __float_as_int(x), 0x4E, 0xF, 0xF, true);
  x = fmaxf(x, __int_as_float(t));   // lane ^ 2
  t = __builtin_amdgcn_update_dpp(0, __float_as_int(x), 0x141, 0xF, 0xF, true);
  x = fmaxf(x, __int_as_float(t));   // ROW_HALF_MIRROR: lane ^ 7 (== ^4 here)
  t = __builtin_amdgcn_update_dpp(0, __float_as_int(x), 0x140, 0xF, 0xF, true);
  x = fmaxf(x, __int_as_float(t));   // ROW_MIRROR: lane ^ 15 (== ^8 here)
  return x;
#else
#pragma unroll
  for (int msk = 8; msk >= 1; msk >>= 1) x = fmaxf(x, __shfl_xor(x, msk, 32));
  return x;
#endif
}

// CDNA5 async global->LDS 16-byte copy (ASYNCcnt-tracked, ISA 15.18.3 op 98).
// Generic LDS pointers truncate to the LDS byte address (ISA 10.2 aperture).
__device__ __forceinline__ void async_copy_b128(void* lds, const void* gptr) {
  const unsigned lds_addr = (unsigned)(unsigned long long)lds;
  asm volatile("global_load_async_to_lds_b128 %0, %1, off"
               :: "v"(lds_addr), "v"(gptr) : "memory");
}
__device__ __forceinline__ void async_wait0() {
  asm volatile("s_wait_asynccnt 0" ::: "memory");
}

// ---------------------------------------------------------------------------
// One-time fp32 -> f16 conversion (vectorized: 8 elements/thread).
// ---------------------------------------------------------------------------
__global__ __launch_bounds__(256) void cvt_f32_to_f16_kernel(
    const float* __restrict__ in, _Float16* __restrict__ out, int n8) {
  const int i = blockIdx.x * blockDim.x + threadIdx.x;
  if (i >= n8) return;
  const float4 f0 = ((const float4*)in)[2 * i + 0];
  const float4 f1 = ((const float4*)in)[2 * i + 1];
  v8h h;
  h[0] = (_Float16)f0.x; h[1] = (_Float16)f0.y;
  h[2] = (_Float16)f0.z; h[3] = (_Float16)f0.w;
  h[4] = (_Float16)f1.x; h[5] = (_Float16)f1.y;
  h[6] = (_Float16)f1.z; h[7] = (_Float16)f1.w;
  *((v8h*)(out + (size_t)i * 8)) = h;
}

// ---------------------------------------------------------------------------
// Wave-level GEMM, all-f16 operands: Out[m,n] = sum_k A[m,k]*W[n,k] + bias[n]
// Each wave computes a 32x64 tile: 2 A-subtiles x 4 B operands = 8 WMMA/kstep.
// OUT_MODE 0: f16 head-major [B,H,S,HD]
// OUT_MODE 1: f16 transposed head-major [B,H,HD,S]   (for V)
// OUT_MODE 2: f32 row-major [M,N]
// ---------------------------------------------------------------------------
template <int OUT_MODE>
__global__ __launch_bounds__(256) void gemm_wmma_kernel(
    const _Float16* __restrict__ A, const _Float16* __restrict__ W,
    const float* __restrict__ bias, void* __restrict__ Out, int N, int K) {
  const int lane   = threadIdx.x & 31;
  const int wave   = threadIdx.x >> 5;
  const int nstrips = N >> 6;                  // 64-wide strips
  const int task   = blockIdx.x * 8 + wave;
  const int mtile  = task / nstrips;           // 32-row tiles
  const int nstrip = task - mtile * nstrips;
  if (mtile * 32 >= kM) return;

  const int l15   = lane & 15;
  const int akoff = (lane >= 16) ? 16 : 0;     // A-operand K-half offset
  const int bkoff = (lane >= 16) ? 16 : 0;     // B-operand K-half offset
  const int m0    = mtile * 32;
  const int nbase = nstrip * 64;

  v8f acc[2][4];
#pragma unroll
  for (int s = 0; s < 2; ++s)
#pragma unroll
    for (int j = 0; j < 4; ++j)
#pragma unroll
      for (int i = 0; i < 8; ++i) acc[s][j][i] = 0.0f;

  for (int kk = 0; kk < K; kk += 32) {
    v16h a[2];
#pragma unroll
    for (int s = 0; s < 2; ++s) {
      const _Float16* p = A + (size_t)(m0 + s * 16 + l15) * K + kk + akoff;
#pragma unroll
      for (int i = 0; i < 8; ++i) { a[s][i] = p[i]; a[s][8 + i] = p[8 + i]; }
      if (kk + 32 < K) __builtin_prefetch(p + 32, 0, 1);
    }
#pragma unroll
    for (int j = 0; j < 4; ++j) {
      const _Float16* wp = W + (size_t)(nbase + j * 16 + l15) * K + kk + bkoff;
      v16h bm;
#pragma unroll
      for (int i = 0; i < 16; ++i) bm[i] = wp[i];
      acc[0][j] = wmma_f16(a[0], bm, acc[0][j]);
      acc[1][j] = wmma_f16(a[1], bm, acc[1][j]);
    }
  }

  // epilogue: C/D layout row = r + (lane>=16 ? 8 : 0)
#pragma unroll
  for (int s = 0; s < 2; ++s) {
    const int row0 = m0 + s * 16 + ((lane >= 16) ? 8 : 0);
#pragma unroll
    for (int j = 0; j < 4; ++j) {
      const int n  = nbase + j * 16 + l15;
      const float bb = bias[n];
#pragma unroll
      for (int r = 0; r < 8; ++r) {
        const int   mm  = row0 + r;
        const float val = acc[s][j][r] + bb;
        if (OUT_MODE == 0) {
          const int b = mm / kS, sr = mm % kS;
          const int h = n / kHD, hd = n % kHD;
          ((_Float16*)Out)[(((size_t)b * kH + h) * kS + sr) * kHD + hd] =
              (_Float16)val;
        } else if (OUT_MODE == 1) {
          const int b = mm / kS, sr = mm % kS;
          const int h = n / kHD, hd = n % kHD;
          ((_Float16*)Out)[(((size_t)b * kH + h) * kHD + hd) * kS + sr] =
              (_Float16)val;
        } else {
          ((float*)Out)[(size_t)mm * N + n] = val;
        }
      }
    }
  }
}

// ---------------------------------------------------------------------------
// Block-cooperative causal flash attention, transposed-output form.
// Block = 8 waves = one (b,h) and 128 consecutive query rows (16 per wave).
// Per 32-key iteration:
//   K (row-major) and V^T ([HD,S]) tiles async-staged into LDS (ASYNCcnt),
//   scores S = Qs K^T via 4 WMMA (scale folded into Q),
//   causal mask only on diagonal-straddling tiles, DPP row-max softmax,
//   P written once to per-wave LDS; then O^T += V^T P via 4 WMMA where
//   A = V^T chunks (contiguous LDS b128 reads) and B = P^T (2 b128 reads),
//   denominator via a 5th WMMA with an all-ones A operand (per-lane result,
//   no end-of-loop shuffles). Alpha crosses layouts via a 16-float per-wave LDS bounce.
// ---------------------------------------------------------------------------
__global__ __launch_bounds__(256) void attn_flash_kernel(
    const _Float16* __restrict__ Qh, const _Float16* __restrict__ Kh,
    const _Float16* __restrict__ VhT, _Float16* __restrict__ O) {
  __shared__ _Float16 Kt[32 * kHD];      // 4 KB, [key][hd]
  __shared__ _Float16 VtT[kHD * 32];     // 4 KB, [hd][key]
  __shared__ _Float16 pbuf[8][16 * 32];  // 8 KB, per-wave P  [qrow][key]
  __shared__ float    abuf[8][16];       // per-wave alpha, indexed by qrow

  const int tid     = threadIdx.x;
  const int lane    = tid & 31;
  const int wave    = tid >> 5;
  const int qchunks = kS / 128;          // 16 blocks of 128 q-rows per (b,h)
  const int bh = blockIdx.x / qchunks;   // b*kH + h
  const int qc = blockIdx.x % qchunks;
  const int h  = bh % kH;
  const int b  = bh / kH;

  const _Float16* Qb = Qh  + (size_t)bh * kS * kHD;
  const _Float16* Kb = Kh  + (size_t)bh * kS * kHD;
  const _Float16* Vb = VhT + (size_t)bh * kHD * kS;  // [hd][s]

  const int qbase   = qc * 128 + wave * 16;
  const int l15     = lane & 15;
  const int akoff   = (lane >= 16) ? 16 : 0;  // A-operand K-half offset
  const int halfoff = (lane >= 16) ? 8 : 0;   // C/D row-half offset
  const int bkoff   = (lane >= 16) ? 16 : 0;  // B-operand K-half offset

  // Q tile as two A operands, softmax scale 0.125 folded in (exact in f16).
  const _Float16 qs = (_Float16)0.125f;
  v16h aq[2];
#pragma unroll
  for (int c = 0; c < 2; ++c) {
    const _Float16* p = Qb + (size_t)(qbase + l15) * kHD + c * 32 + akoff;
#pragma unroll
    for (int i = 0; i < 8; ++i) {
      aq[c][i]     = p[i] * qs;
      aq[c][8 + i] = p[8 + i] * qs;
    }
  }

  // all-ones A operand: row-sum (softmax denominator) WMMA
  v16h aones;
#pragma unroll
  for (int i = 0; i < 16; ++i) aones[i] = (_Float16)1.0f;

  v8f o[4], osum;   // o[c] = O^T chunk (rows=hd, cols=qrow); osum = denoms
#pragma unroll
  for (int c = 0; c < 4; ++c)
#pragma unroll
    for (int i = 0; i < 8; ++i) o[c][i] = 0.0f;
#pragma unroll
  for (int i = 0; i < 8; ++i) osum[i] = 0.0f;
  float mi[8];
#pragma unroll
  for (int r = 0; r < 8; ++r) mi[r] = kNegInf;

  const int kend_w   = qbase + 16;      // this wave's causal limit
  const int kend_blk = qc * 128 + 128;  // block's causal limit

  const int srow = tid >> 3;            // K staging: key row 0..31
  const int scol = (tid & 7) * 8;       // K staging: hd col 0,8,..,56
  const int vrow = tid >> 2;            // V^T staging: hd row 0..63
  const int vcol = (tid & 3) * 8;       // V^T staging: key col 0,8,16,24

  for (int kb = 0; kb < kend_blk; kb += 32) {
    // ---- async K / V^T tile staging: one b128 per thread per tile --------
    async_copy_b128(Kt + srow * kHD + scol,
                    Kb + (size_t)(kb + srow) * kHD + scol);
    async_copy_b128(VtT + vrow * 32 + vcol,
                    Vb + (size_t)vrow * kS + kb + vcol);
    async_wait0();
    __syncthreads();

    if (kb < kend_w) {
      // ---------- scores: two 16x16 tiles over 32 keys (from LDS K) --------
      v8f sc[2];
#pragma unroll
      for (int t = 0; t < 2; ++t)
#pragma unroll
        for (int i = 0; i < 8; ++i) sc[t][i] = 0.0f;

#pragma unroll
      for (int t = 0; t < 2; ++t) {
        const int kv = t * 16 + l15;  // local key row = B-operand column
#pragma unroll
        for (int c = 0; c < 2; ++c) {
          const _Float16* p = Kt + kv * kHD + c * 32 + bkoff;
          v16h bk;
#pragma unroll
          for (int i = 0; i < 16; ++i) bk[i] = p[i];
          sc[t] = wmma_f16(aq[c], bk, sc[t]);
        }
      }

      // ---------- causal mask: only tiles straddling the diagonal ----------
      if (kb + 31 > qbase) {  // wave-uniform; <=2 iterations per wave
#pragma unroll
        for (int t = 0; t < 2; ++t) {
          const int key = kb + t * 16 + l15;
#pragma unroll
          for (int r = 0; r < 8; ++r) {
            const int qr = qbase + halfoff + r;
            sc[t][r] = (key <= qr) ? sc[t][r] : kNegInf;
          }
        }
      }

      // ---------- online softmax: DPP row-max over 16-lane halves ----------
      float alpha[8];
#pragma unroll
      for (int r = 0; r < 8; ++r) {
        const float rm = half16_max(fmaxf(sc[0][r], sc[1][r]));
        const float nm = fmaxf(mi[r], rm);
        alpha[r] = __expf(mi[r] - nm);
        mi[r] = nm;
        sc[0][r] = __expf(sc[0][r] - nm);
        sc[1][r] = __expf(sc[1][r] - nm);
      }

      // ---------- P -> per-wave LDS (row-major by qrow); alpha bounce ------
      _Float16* pb = pbuf[wave];
#pragma unroll
      for (int t = 0; t < 2; ++t)
#pragma unroll
        for (int r = 0; r < 8; ++r)
          pb[(halfoff + r) * 32 + t * 16 + l15] = (_Float16)sc[t][r];
      if (l15 == 0) {  // lanes 0 and 16 hold rows 0-7 / 8-15 respectively
#pragma unroll
        for (int r = 0; r < 8; ++r) abuf[wave][halfoff + r] = alpha[r];
      }

      asm volatile("s_wait_dscnt 0" ::: "memory");  // wave-local LDS RAW fence

      // B operand = P^T: column = qrow(l15), K = keys (contiguous in pbuf)
      v16h bp;
      {
        const _Float16* pr = pb + l15 * 32 + bkoff;
#pragma unroll
        for (int i = 0; i < 16; ++i) bp[i] = pr[i];
      }
      // rescale accumulators by this lane's q-row alpha
      const float alane = abuf[wave][l15];
#pragma unroll
      for (int c = 0; c < 4; ++c)
#pragma unroll
        for (int i = 0; i < 8; ++i) o[c][i] *= alane;
#pragma unroll
      for (int i = 0; i < 8; ++i) osum[i] *= alane;

      // ---------- O^T += V^T x P : A = V^T chunks (contiguous LDS) ---------
#pragma unroll
      for (int c = 0; c < 4; ++c) {
        const _Float16* vp = VtT + (c * 16 + l15) * 32 + akoff;
        v16h av;
#pragma unroll
        for (int i = 0; i < 8; ++i) { av[i] = vp[i]; av[8 + i] = vp[8 + i]; }
        o[c] = wmma_f16(av, bp, o[c]);
      }
      osum = wmma_f16(aones, bp, osum);  // denominators, replicated per lane
    }
    __syncthreads();  // protect LDS tiles before next staging
  }

  // ---- normalize & store to [B, S, H*HD] f16 (per-lane denominator) ----
  const float inv = 1.0f / osum[0];
  const int   qr  = qbase + l15;
#pragma unroll
  for (int c = 0; c < 4; ++c)
#pragma unroll
    for (int r = 0; r < 8; ++r) {
      const int hd = c * 16 + halfoff + r;
      O[((size_t)b * kS + qr) * (kH * kHD) + h * kHD + hd] =
          (_Float16)(o[c][r] * inv);
    }
}

// ---------------------------------------------------------------------------
extern "C" void kernel_launch(void* const* d_in, const int* in_sizes, int n_in,
                              void* d_out, int out_size, void* d_ws,
                              size_t ws_size, hipStream_t stream) {
  (void)in_sizes; (void)n_in; (void)out_size; (void)ws_size;
  const float* q  = (const float*)d_in[0];
  const float* k  = (const float*)d_in[1];
  const float* v  = (const float*)d_in[2];
  const float* Wq = (const float*)d_in[3];
  const float* bq = (const float*)d_in[4];
  const float* Wk = (const float*)d_in[5];
  const float* bk = (const float*)d_in[6];
  const float* Wv = (const float*)d_in[7];
  const float* bv = (const float*)d_in[8];
  const float* Wo = (const float*)d_in[9];
  const float* bo = (const float*)d_in[10];
  float* out = (float*)d_out;

  // f16 workspace layout
  const size_t nAct = (size_t)kB * kS * kD;  // 4,194,304
  const size_t nW   = (size_t)kD * kD;       // 1,048,576
  _Float16* qf  = (_Float16*)d_ws;           // activations (f16)
  _Float16* kf  = qf + nAct;
  _Float16* vf  = kf + nAct;
  _Float16* Wqh = vf + nAct;                 // weights (f16)
  _Float16* Wkh = Wqh + nW;
  _Float16* Wvh = Wkh + nW;
  _Float16* Woh = Wvh + nW;
  _Float16* Qh  = Woh + nW;                  // head-major Q [B,H,S,HD]
  _Float16* Kh  = Qh + nAct;                 // head-major K [B,H,S,HD]
  _Float16* VhT = Kh + nAct;                 // transposed V [B,H,HD,S]
  _Float16* attn = VhT + nAct;               // attention output [B,S,H*HD]

  const dim3 blk(256);

  // one-time fp32 -> f16 conversions
  const int gAct = (int)(nAct / 8 / 256);
  const int gW   = (int)(nW / 8 / 256);
  cvt_f32_to_f16_kernel<<<gAct, blk, 0, stream>>>(q, qf, (int)(nAct / 8));
  cvt_f32_to_f16_kernel<<<gAct, blk, 0, stream>>>(k, kf, (int)(nAct / 8));
  cvt_f32_to_f16_kernel<<<gAct, blk, 0, stream>>>(v, vf, (int)(nAct / 8));
  cvt_f32_to_f16_kernel<<<gW, blk, 0, stream>>>(Wq, Wqh, (int)(nW / 8));
  cvt_f32_to_f16_kernel<<<gW, blk, 0, stream>>>(Wk, Wkh, (int)(nW / 8));
  cvt_f32_to_f16_kernel<<<gW, blk, 0, stream>>>(Wv, Wvh, (int)(nW / 8));
  cvt_f32_to_f16_kernel<<<gW, blk, 0, stream>>>(Wo, Woh, (int)(nW / 8));

  // QKV projections: f16 GEMM -> f16 head-major (V transposed)
  const int gemm_tasks = (kM / 32) * (kD / 64);  // 2048 wave tiles
  const dim3 ggrd(gemm_tasks / 8);
  gemm_wmma_kernel<0><<<ggrd, blk, 0, stream>>>(qf, Wqh, bq, Qh, kD, kD);
  gemm_wmma_kernel<0><<<ggrd, blk, 0, stream>>>(kf, Wkh, bk, Kh, kD, kD);
  gemm_wmma_kernel<1><<<ggrd, blk, 0, stream>>>(vf, Wvh, bv, VhT, kD, kD);

  // causal flash attention: one block per (b, h, 128-row q chunk)
  const int attn_blocks = kB * kH * (kS / 128);  // 512
  attn_flash_kernel<<<dim3(attn_blocks), blk, 0, stream>>>(Qh, Kh, VhT, attn);

  // output projection: f16 GEMM -> fp32 row-major (+bo)
  gemm_wmma_kernel<2><<<ggrd, blk, 0, stream>>>(attn, Woh, bo, out, kD, kD);
}
